// KnowledgeGCN_54966991454756
// MI455X (gfx1250) — matrix operand order repaired and checked by reading, not compiled, not run on hardware
//
#include <hip/hip_runtime.h>
#include <hip/hip_bf16.h>
#include <math.h>

#define DEV __device__ __forceinline__
typedef unsigned short u16;

namespace {

constexpr int Bb   = 128;   // batch
constexpr int NKG  = 500;   // KG vertices
constexpr int Nn   = 502;   // + sensor + zero node
constexpr int NP   = 512;   // padded node count (rows/cols of dense A)
constexpr int Dd   = 1024;  // embed dim
constexpr int Hh   = 256;   // hidden
constexpr int Ee   = 4000;  // edges
constexpr int MC   = 640;   // padded 628 = 500 base + 128 sensor rows
constexpr float EPSBN = 1e-5f;

// LDS layout for the A slab: 16 rows x 512 f32, 2 pad dwords every 256 dwords
// -> effective row stride 516 dwords, bank-conflict-free WMMA fragment reads.
constexpr int LDS_A_DWORDS = 16 * 512 + (16 * 512 / 256) * 2;   // 8256

typedef __attribute__((ext_vector_type(8))) float v8f;

#if __has_builtin(__builtin_amdgcn_wmma_f32_16x16x32_bf16)
#define HAVE_WMMA_BF16 1
typedef __attribute__((ext_vector_type(16))) __bf16 v16bf;
struct U32x8 { uint4 a, b; };
#endif
#if __has_builtin(__builtin_amdgcn_wmma_f32_16x16x4_f32)
#define HAVE_WMMA_F32 1
#endif
typedef __attribute__((ext_vector_type(2))) float v2f;
#if __has_builtin(__builtin_amdgcn_tensor_load_to_lds)
#define HAVE_TDM 1
typedef __attribute__((ext_vector_type(4))) unsigned int v4u;
typedef __attribute__((ext_vector_type(8))) int v8i;
typedef __attribute__((ext_vector_type(4))) int v4i;
#endif

DEV float bf2f(u16 u) { unsigned int x = ((unsigned int)u) << 16; return __builtin_bit_cast(float, x); }
DEV u16 f2bf(float f) { return __builtin_bit_cast(u16, __float2bfloat16(f)); }
DEV int ldsIdx(int flat) { return flat + ((flat >> 8) << 1); }   // padded LDS address

// ---------------------------------------------------------------- graph setup
__global__ void kZeroInit(float* __restrict__ A, float* __restrict__ deg) {
  int i = blockIdx.x * blockDim.x + threadIdx.x;
  if (i < NP * NP) A[i] = 0.0f;
  if (i < NP) deg[i] = (i < Nn) ? 1.0f : 0.0f;   // self-loop pre-count
}

__global__ void kDeg(const int* __restrict__ ei, float* __restrict__ deg) {
  int e = blockIdx.x * blockDim.x + threadIdx.x;
  if (e < Ee) atomicAdd(&deg[ei[Ee + e]], 1.0f);
}

__global__ void kDinv(const float* __restrict__ deg, float* __restrict__ dinv) {
  int n = blockIdx.x * blockDim.x + threadIdx.x;
  if (n < Nn) dinv[n] = rsqrtf(deg[n]);
}

__global__ void kFillA(const int* __restrict__ ei, const float* __restrict__ dinv,
                       float* __restrict__ A) {
  int i = blockIdx.x * blockDim.x + threadIdx.x;
  if (i < Ee) {
    int s = ei[i], d = ei[Ee + i];
    atomicAdd(&A[d * NP + s], dinv[s] * dinv[d]);
  } else if (i < Ee + Nn) {
    int n = i - Ee;
    atomicAdd(&A[n * NP + n], dinv[n] * dinv[n]);
  }
}

// ------------------------------------------------- input packing / transposes
__global__ void kCastX1(const float* __restrict__ base, const float* __restrict__ sensor,
                        u16* __restrict__ Xcat) {
  int i = blockIdx.x * blockDim.x + threadIdx.x;
  if (i >= MC * Dd) return;
  int r = i / Dd, c = i - r * Dd;
  float v = 0.0f;
  if (r < NKG)            v = base[r * Dd + c];
  else if (r < NKG + Bb)  v = sensor[(r - NKG) * Dd + c];
  Xcat[i] = f2bf(v);
}

// WT[h][k] = bf16(W[k][h])
__global__ void kTransW(const float* __restrict__ W, u16* __restrict__ WT, int K, int Hc) {
  int i = blockIdx.x * blockDim.x + threadIdx.x;
  if (i >= K * Hc) return;
  int h = i / K, k = i - h * K;
  WT[i] = f2bf(W[k * Hc + h]);
}

// -------------------------------------------------------- bf16 WMMA GEMM
// C[M,Hh] (f32) = X[M,K](bf16) * W, WT = W^T as [Hh][K] bf16.
__global__ void __launch_bounds__(256) kGemmBf16(const u16* __restrict__ X, int ldx,
                                                 const u16* __restrict__ WT,
                                                 float* __restrict__ C,
                                                 int M, int K) {
  int wave = (blockIdx.x * blockDim.x + threadIdx.x) >> 5;
  int lane = threadIdx.x & 31;
  const int ntiles = Hh >> 4;
  int mt = wave / ntiles, nt = wave - mt * ntiles;
  int m0 = mt << 4, n0 = nt << 4;
  if (m0 >= M) return;                       // wave-uniform; EXEC stays all-ones
  int rbase = m0 + ((lane >> 4) << 3);
  int col   = n0 + (lane & 15);
#if defined(HAVE_WMMA_BF16) && defined(__AMDGCN__)
  int rowA  = m0 + (lane & 15);
  int koffA = (lane >> 4) * 8;               // A: lanes 0-15 K 0..7/16..23, 16-31 K 8..15/24..31
  int koffB = (lane >> 4) * 8;               // B: lanes 0-15 K 0..7/16..23, 16-31 K 8..15/24..31
  const u16* pa = X  + (size_t)rowA * ldx + koffA;
  const u16* pb = WT + (size_t)col  * K   + koffB;
  v8f acc = {};
  for (int k0 = 0; k0 < K; k0 += 32) {
    U32x8 ta, tb;
    ta.a = *(const uint4*)(pa + k0);
    ta.b = *(const uint4*)(pa + k0 + 16);
    tb.a = *(const uint4*)(pb + k0);
    tb.b = *(const uint4*)(pb + k0 + 16);
    v16bf fa = __builtin_bit_cast(v16bf, ta);
    v16bf fb = __builtin_bit_cast(v16bf, tb);
    acc = __builtin_amdgcn_wmma_f32_16x16x32_bf16(false, fa, false, fb,
                                                  (short)0, acc, false, false);
  }
#pragma unroll
  for (int r = 0; r < 8; ++r) C[(size_t)(rbase + r) * Hh + col] = acc[r];
#else
  float acc[8] = {};
  for (int k = 0; k < K; ++k) {
    float wv = bf2f(WT[(size_t)col * K + k]);
#pragma unroll
    for (int r = 0; r < 8; ++r) acc[r] += bf2f(X[(size_t)(rbase + r) * ldx + k]) * wv;
  }
#pragma unroll
  for (int r = 0; r < 8; ++r) C[(size_t)(rbase + r) * Hh + col] = acc[r];
#endif
}

// ---------------------------------------------- expand layer-1 hidden to batch
__global__ void kExpandH1(const float* __restrict__ hsmall, float* __restrict__ hfull) {
  size_t i = (size_t)blockIdx.x * blockDim.x + threadIdx.x;
  if (i >= (size_t)Bb * NP * Hh) return;
  int f = i & (Hh - 1);
  size_t rn = i >> 8;               // /Hh
  int n = rn & (NP - 1);
  int b = (int)(rn >> 9);           // /NP
  float v = 0.0f;
  if (n < NKG)       v = hsmall[(size_t)n * Hh + f];
  else if (n == NKG) v = hsmall[(size_t)(NKG + b) * Hh + f];
  hfull[i] = v;
}

// -------------------------------------------------- f32 WMMA aggregation GEMM
// Aout[b] = A[NP,NP] @ H[b][NP,Hh].  One block per (b, row-tile): the 16x512
// A slab (32KB) is DMA'd into LDS once by the Tensor Data Mover and shared by
// all 8 waves (one wave = two 16-col tiles).  Full f32 via v_wmma_f32_16x16x4.
__global__ void __launch_bounds__(256) kAggF32(const float* __restrict__ A,
                                               const float* __restrict__ Hm,
                                               float* __restrict__ Aout) {
  __shared__ float Ald[LDS_A_DWORDS];
  int mt = blockIdx.x & 31;
  int b  = blockIdx.x >> 5;
  int m0 = mt << 4;

#if defined(HAVE_TDM) && defined(__AMDGCN__)
  if (threadIdx.x < 32) {                      // wave 0 issues the DMA
    unsigned long long ga =
        (unsigned long long)(uintptr_t)(A + (size_t)m0 * NP);
    // D# group0: count=1 | lds_addr=0 | global_addr | type=2 (bits 127:126)
    v4u g0 = { 1u, 0u, (unsigned)(ga & 0xffffffffull),
               (unsigned)(((ga >> 32) & 0x1ffffffull) | 0x80000000u) };
    // D# group1: data_size=4B(2), pad_enable, pad_interval=256dw(7),
    //            pad_amount=2dw(1); tensor_dim0 = tile_dim0 = 8192 f32 (1-D)
    v8i g1 = { (int)0x03D20000u,        // mask=0, ds=2, pad en/intv/amt
               (int)0x20000000u,        // tensor_dim0[15:0]=8192 in bits 63:48
               0,                       // tensor_dim0[31:16]=0, tensor_dim1 lo=0
               (int)0x20000000u,        // tile_dim0=8192 in bits 127:112
               0,                       // tile_dim1=0 (unused), tile_dim2=0
               8192,                    // tensor_dim0_stride lo
               0, 0 };
    v4i g2 = { 0, 0, 0, 0 };
    v4i g3 = { 0, 0, 0, 0 };
    v8i g4 = { 0, 0, 0, 0, 0, 0, 0, 0 };   // 6-arg toolchain: extra group, unused
    __builtin_amdgcn_tensor_load_to_lds(g0, g1, g2, g3, g4, 0);
    __builtin_amdgcn_s_wait_tensorcnt((short)0);
  }
  // The TDM builtin carries no pointer operand, so LLVM cannot see that it
  // wrote Ald.  Escape the pointer through an asm memory clobber so the
  // subsequent ds_load fragment reads are not optimized away.
  {
    void* esc = (void*)Ald;
    asm volatile("" : "+v"(esc) : : "memory");
  }
#else
  for (int i = threadIdx.x; i < 16 * NP; i += 256)
    Ald[ldsIdx(i)] = A[(size_t)m0 * NP + i];
#endif
  __syncthreads();

  int wave = threadIdx.x >> 5;
  int lane = threadIdx.x & 31;
  const float* hb = Hm + (size_t)b * NP * Hh;
  float* ab = Aout + (size_t)b * NP * Hh;
  int rl = lane & 15;
  int ka = (lane >> 4) * 2;                    // half-wave K split {0,1}/{2,3}
  int rbase = m0 + ((lane >> 4) << 3);

#pragma unroll
  for (int t = 0; t < 2; ++t) {
    int n0 = ((wave << 1) | t) << 4;
    int col = n0 + (lane & 15);
#if defined(HAVE_WMMA_F32) && defined(__AMDGCN__)
    v8f acc = {};
    for (int k0 = 0; k0 < NP; k0 += 4) {
      int k = k0 + ka;
      int la = (rl << 9) + (rl << 2) + k + ((k >> 8) << 1);   // 516*rl + pad(k)
      v2f fa; fa.x = Ald[la]; fa.y = Ald[la + 1];
      v2f fb;
      fb.x = hb[(size_t)k * Hh + col];
      fb.y = hb[(size_t)(k + 1) * Hh + col];
      acc = __builtin_amdgcn_wmma_f32_16x16x4_f32(false, fa, false, fb,
                                                  (short)0, acc, false, false);
    }
#pragma unroll
    for (int r = 0; r < 8; ++r) ab[(size_t)(rbase + r) * Hh + col] = acc[r];
#else
    float acc[8] = {};
    for (int k = 0; k < NP; ++k) {
      float hv = hb[(size_t)k * Hh + col];
#pragma unroll
      for (int r = 0; r < 8; ++r)
        acc[r] += Ald[ldsIdx(((rbase + r - m0) << 9) + k)] * hv;
    }
#pragma unroll
    for (int r = 0; r < 8; ++r) ab[(size_t)(rbase + r) * Hh + col] = acc[r];
#endif
  }
}

// ----------------------------------------------------------- BatchNorm + ELU
__global__ void kZeroStats(float* __restrict__ s0, float* __restrict__ s1) {
  int i = threadIdx.x;
  if (i < Hh) { s0[i] = 0.0f; s1[i] = 0.0f; }
}

__global__ void kStats(const float* __restrict__ Aout, float* __restrict__ ssum,
                       float* __restrict__ ssq) {
  int f = threadIdx.x;                    // 256 features
  float s = 0.0f, q = 0.0f;
  for (int r = blockIdx.x; r < Bb * Nn; r += gridDim.x) {
    int b = r / Nn, n = r - b * Nn;
    float v = Aout[((size_t)b * NP + n) * Hh + f];
    s += v; q += v * v;
  }
  atomicAdd(&ssum[f], s);
  atomicAdd(&ssq[f], q);
}

__global__ void kFinalize(const float* __restrict__ ssum, const float* __restrict__ ssq,
                          const float* __restrict__ gamma, const float* __restrict__ beta,
                          float* __restrict__ scale, float* __restrict__ shift) {
  int f = threadIdx.x;
  const float inv = 1.0f / (float)(Bb * Nn);
  float m = ssum[f] * inv;
  float var = ssq[f] * inv - m * m;      // bias term cancels in training-mode BN
  float sc = gamma[f] * rsqrtf(var + EPSBN);
  scale[f] = sc;
  shift[f] = beta[f] - m * sc;
}

__global__ void kApplyBnElu(const float* __restrict__ Aout,
                            const float* __restrict__ scale, const float* __restrict__ shift,
                            u16* __restrict__ xnext, float* __restrict__ outf) {
  size_t i = (size_t)blockIdx.x * blockDim.x + threadIdx.x;
  if (i >= (size_t)Bb * NP * Hh) return;
  int f = i & (Hh - 1);
  size_t rn = i >> 8;
  int n = rn & (NP - 1);
  int b = (int)(rn >> 9);
  if (n < Nn) {
    float y = Aout[i] * scale[f] + shift[f];
    y = (y > 0.0f) ? y : expm1f(y);
    if (xnext) xnext[i] = f2bf(y);
    if (outf)  outf[((size_t)b * Nn + n) * Hh + f] = y;
  } else if (xnext) {
    xnext[i] = 0;                        // keep padded rows exactly zero
  }
}

} // namespace

// ------------------------------------------------------------------- launcher
extern "C" void kernel_launch(void* const* d_in, const int* in_sizes, int n_in,
                              void* d_out, int out_size, void* d_ws, size_t ws_size,
                              hipStream_t stream) {
  const float* sensor = (const float*)d_in[0];
  const float* base   = (const float*)d_in[1];
  const int*   ei     = (const int*)d_in[2];
  const float* W1 = (const float*)d_in[3];
  const float* g1 = (const float*)d_in[5];
  const float* be1 = (const float*)d_in[6];
  const float* W2 = (const float*)d_in[7];
  const float* g2 = (const float*)d_in[9];
  const float* be2 = (const float*)d_in[10];
  const float* W3 = (const float*)d_in[11];
  const float* g3 = (const float*)d_in[13];
  const float* be3 = (const float*)d_in[14];
  float* out = (float*)d_out;
  (void)in_sizes; (void)n_in; (void)out_size; (void)ws_size;

  char* ws = (char*)d_ws;
  size_t off = 0;
  auto alloc = [&](size_t bytes) -> char* {
    char* p = ws + off;
    off = (off + bytes + 255) & ~(size_t)255;
    return p;
  };
  float* A      = (float*)alloc((size_t)NP * NP * 4);
  float* deg    = (float*)alloc(NP * 4);
  float* dinv   = (float*)alloc(NP * 4);
  u16*   Xcat   = (u16*)  alloc((size_t)MC * Dd * 2);
  u16*   WT1    = (u16*)  alloc((size_t)Hh * Dd * 2);
  u16*   WT2    = (u16*)  alloc((size_t)Hh * Hh * 2);
  u16*   WT3    = (u16*)  alloc((size_t)Hh * Hh * 2);
  float* hsmall = (float*)alloc((size_t)MC * Hh * 4);
  float* hfull  = (float*)alloc((size_t)Bb * NP * Hh * 4);
  float* afull  = (float*)alloc((size_t)Bb * NP * Hh * 4);
  u16*   xbf    = (u16*)  alloc((size_t)Bb * NP * Hh * 2);
  float* ssum   = (float*)alloc(Hh * 4);
  float* ssq    = (float*)alloc(Hh * 4);
  float* scale  = (float*)alloc(Hh * 4);
  float* shift  = (float*)alloc(Hh * 4);

  const int TB = 256;
  auto cdiv = [](long long a, long long b) -> int { return (int)((a + b - 1) / b); };

  // ---- normalized dense adjacency
  kZeroInit<<<cdiv(NP * NP, TB), TB, 0, stream>>>(A, deg);
  kDeg<<<cdiv(Ee, TB), TB, 0, stream>>>(ei, deg);
  kDinv<<<cdiv(Nn, TB), TB, 0, stream>>>(deg, dinv);
  kFillA<<<cdiv(Ee + Nn, TB), TB, 0, stream>>>(ei, dinv, A);

  // ---- pack inputs / transpose weights to bf16
  kCastX1<<<cdiv((size_t)MC * Dd, TB), TB, 0, stream>>>(base, sensor, Xcat);
  kTransW<<<cdiv((size_t)Dd * Hh, TB), TB, 0, stream>>>(W1, WT1, Dd, Hh);
  kTransW<<<cdiv((size_t)Hh * Hh, TB), TB, 0, stream>>>(W2, WT2, Hh, Hh);
  kTransW<<<cdiv((size_t)Hh * Hh, TB), TB, 0, stream>>>(W3, WT3, Hh, Hh);

  const long long totalBNH = (long long)Bb * NP * Hh;
  const int aggBlocks = Bb * 32;            // one block per (b, row-tile)

  // ---- layer 1 (structure-exploiting): hsmall = [base;sensor] @ W1
  kGemmBf16<<<cdiv((long long)(MC / 16) * (Hh / 16) * 32, TB), TB, 0, stream>>>(
      Xcat, Dd, WT1, hsmall, MC, Dd);
  kExpandH1<<<cdiv(totalBNH, TB), TB, 0, stream>>>(hsmall, hfull);
  kAggF32<<<aggBlocks, TB, 0, stream>>>(A, hfull, afull);
  kZeroStats<<<1, TB, 0, stream>>>(ssum, ssq);
  kStats<<<512, TB, 0, stream>>>(afull, ssum, ssq);
  kFinalize<<<1, TB, 0, stream>>>(ssum, ssq, g1, be1, scale, shift);
  kApplyBnElu<<<cdiv(totalBNH, TB), TB, 0, stream>>>(afull, scale, shift, xbf, nullptr);

  // ---- layer 2
  kGemmBf16<<<cdiv((long long)(Bb * NP / 16) * (Hh / 16) * 32, TB), TB, 0, stream>>>(
      xbf, Hh, WT2, hfull, Bb * NP, Hh);
  kAggF32<<<aggBlocks, TB, 0, stream>>>(A, hfull, afull);
  kZeroStats<<<1, TB, 0, stream>>>(ssum, ssq);
  kStats<<<512, TB, 0, stream>>>(afull, ssum, ssq);
  kFinalize<<<1, TB, 0, stream>>>(ssum, ssq, g2, be2, scale, shift);
  kApplyBnElu<<<cdiv(totalBNH, TB), TB, 0, stream>>>(afull, scale, shift, xbf, nullptr);

  // ---- layer 3 (final output, packed [B, 502, 256] f32)
  kGemmBf16<<<cdiv((long long)(Bb * NP / 16) * (Hh / 16) * 32, TB), TB, 0, stream>>>(
      xbf, Hh, WT3, hfull, Bb * NP, Hh);
  kAggF32<<<aggBlocks, TB, 0, stream>>>(A, hfull, afull);
  kZeroStats<<<1, TB, 0, stream>>>(ssum, ssq);
  kStats<<<512, TB, 0, stream>>>(afull, ssum, ssq);
  kFinalize<<<1, TB, 0, stream>>>(ssum, ssq, g3, be3, scale, shift);
  kApplyBnElu<<<cdiv(totalBNH, TB), TB, 0, stream>>>(afull, scale, shift, nullptr, out);
}
